// LSTM_74002286510419
// MI455X (gfx1250) — matrix operand order repaired and checked by reading, not compiled, not run on hardware
//
#include <hip/hip_runtime.h>
#include <hip/hip_bf16.h>

#define BSZ   4
#define SEQ   4096
#define EMB   1024
#define HID   2048
#define FOURH 8192
#define NHEAD 16
#define HD    128

#define TM 128
#define TN 128
#define TK 32

typedef __attribute__((ext_vector_type(16))) __bf16 v16bf;
typedef __attribute__((ext_vector_type(8)))  __bf16 v8bf;
typedef __attribute__((ext_vector_type(8)))  float  v8f;

// GCC-style v4i to match the builtin's declared parameter type exactly.
typedef int v4i_gcc __attribute__((vector_size(16)));

#if defined(__gfx1250__) && \
    __has_builtin(__builtin_amdgcn_global_load_async_to_lds_b128) && \
    __has_builtin(__builtin_amdgcn_s_wait_asynccnt)
#define USE_ASYNC 1
#else
#define USE_ASYNC 0
#endif

__device__ __forceinline__ unsigned short bf16_rne(float x) {
    unsigned int u = __float_as_uint(x);
    u += 0x7FFFu + ((u >> 16) & 1u);
    return (unsigned short)(u >> 16);
}

__device__ __forceinline__ float sigmoidf_dev(float x) {
    return 1.0f / (1.0f + __expf(-x));
}

// 16-byte global -> LDS chunk copy (ASYNCcnt-tracked path when available).
__device__ __forceinline__ void copy_chunk(const unsigned short* g, unsigned short* l) {
#if USE_ASYNC
    __builtin_amdgcn_global_load_async_to_lds_b128(
        (__attribute__((address_space(1))) v4i_gcc*)g,
        (__attribute__((address_space(3))) v4i_gcc*)l, 0, 0);
#else
    *(v8bf*)l = *(const v8bf*)g;
#endif
}

// fp32 -> (bf16 hi, bf16 lo) split so the GEMM inner loop is conversion-free.
__global__ void LSTM_split_bf16(const float* __restrict__ in,
                                unsigned short* __restrict__ hi,
                                unsigned short* __restrict__ lo,
                                int n) {
    int stride = gridDim.x * blockDim.x;
    for (int i = blockIdx.x * blockDim.x + threadIdx.x; i < n; i += stride) {
        float x = in[i];
        unsigned short h = bf16_rne(x);
        float fh = __uint_as_float(((unsigned int)h) << 16);
        hi[i] = h;
        lo[i] = bf16_rne(x - fh);
    }
}

// C[M,N] = A[M,K] * B[N,K]^T, split-bf16 WMMA (hi*hi + hi*lo + lo*hi).
// Block tile 128x128, K-step 32, double-buffered LDS staging (async-to-LDS),
// 8 waves of 64x32 each: 24 wmma per wave per K-step.
__global__ __launch_bounds__(256)
void LSTM_gemm_lds(const unsigned short* __restrict__ Ah,
                   const unsigned short* __restrict__ Al,
                   const unsigned short* __restrict__ Bh,
                   const unsigned short* __restrict__ Bl,
                   float* __restrict__ C,
                   int M, int N, int K) {
    __shared__ __align__(16) unsigned short sm[2][4][TM * TK];  // 64 KB

    const int tid  = threadIdx.x;
    const int lane = tid & 31;
    const int w    = tid >> 5;
    const int wm   = w & 1;      // 0..1 -> rows wm*64
    const int wn   = w >> 1;     // 0..3 -> cols wn*32
    const int half = lane >> 4;
    const int r    = lane & 15;

    const int m0b = blockIdx.y * TM;
    const int n0b = blockIdx.x * TN;

    // Staging plan: 4 arrays (Ah,Al,Bh,Bl) x 512 chunks of 16B per K-step.
    // Each thread copies 2 chunks per array per stage (8 async ops/thread/stage).
    const unsigned short* gA[2][2];   // [hi/lo][j]
    const unsigned short* gB[2][2];
    int loff[2];
    #pragma unroll
    for (int j = 0; j < 2; ++j) {
        int c   = tid + j * 256;
        int row = c >> 2;
        int q   = c & 3;
        loff[j] = row * TK + q * 8;
        gA[0][j] = Ah + (long)(m0b + row) * K + q * 8;
        gA[1][j] = Al + (long)(m0b + row) * K + q * 8;
        gB[0][j] = Bh + (long)(n0b + row) * K + q * 8;
        gB[1][j] = Bl + (long)(n0b + row) * K + q * 8;
    }

    v8f c[4][2] = {};

    const int nsteps = K / TK;

    // prologue: stage buffer 0
    #pragma unroll
    for (int j = 0; j < 2; ++j) {
        copy_chunk(gA[0][j], &sm[0][0][loff[j]]);
        copy_chunk(gA[1][j], &sm[0][1][loff[j]]);
        copy_chunk(gB[0][j], &sm[0][2][loff[j]]);
        copy_chunk(gB[1][j], &sm[0][3][loff[j]]);
    }

    for (int s = 0; s < nsteps; ++s) {
        const int buf = s & 1;
        if (s + 1 < nsteps) {
            const int nb = (s + 1) & 1;
            const int k0 = (s + 1) * TK;
            #pragma unroll
            for (int j = 0; j < 2; ++j) {
                copy_chunk(gA[0][j] + k0, &sm[nb][0][loff[j]]);
                copy_chunk(gA[1][j] + k0, &sm[nb][1][loff[j]]);
                copy_chunk(gB[0][j] + k0, &sm[nb][2][loff[j]]);
                copy_chunk(gB[1][j] + k0, &sm[nb][3][loff[j]]);
            }
        }
#if USE_ASYNC
        if (s + 1 < nsteps) __builtin_amdgcn_s_wait_asynccnt(8);
        else                __builtin_amdgcn_s_wait_asynccnt(0);
#endif
        __syncthreads();   // stage s visible to all waves

        const unsigned short* sAh = sm[buf][0];
        const unsigned short* sAl = sm[buf][1];
        const unsigned short* sBh = sm[buf][2];
        const unsigned short* sBl = sm[buf][3];

        // B fragments: lane r = column, half selects K 0..15 / 16..31 (32B contig)
        v16bf bh[2], bl[2];
        #pragma unroll
        for (int nt = 0; nt < 2; ++nt) {
            int base = (wn * 32 + nt * 16 + r) * TK + half * 16;
            bh[nt] = *(const v16bf*)(sBh + base);
            bl[nt] = *(const v16bf*)(sBl + base);
        }

        #pragma unroll
        for (int mt = 0; mt < 4; ++mt) {
            int abase = (wm * 64 + mt * 16 + r) * TK;
            v16bf ah = *(const v16bf*)(sAh + abase + half * 16);
            v16bf al = *(const v16bf*)(sAl + abase + half * 16);

            #pragma unroll
            for (int nt = 0; nt < 2; ++nt) {
                c[mt][nt] = __builtin_amdgcn_wmma_f32_16x16x32_bf16(
                    false, ah, false, bh[nt], (short)0, c[mt][nt], false, false);
                c[mt][nt] = __builtin_amdgcn_wmma_f32_16x16x32_bf16(
                    false, ah, false, bl[nt], (short)0, c[mt][nt], false, false);
                c[mt][nt] = __builtin_amdgcn_wmma_f32_16x16x32_bf16(
                    false, al, false, bh[nt], (short)0, c[mt][nt], false, false);
            }
        }
        __syncthreads();   // all reads of buf done before it is restaged
    }

    // C/D layout: VGPR v -> row half*8+v, lane r -> column
    #pragma unroll
    for (int mt = 0; mt < 4; ++mt) {
        #pragma unroll
        for (int nt = 0; nt < 2; ++nt) {
            #pragma unroll
            for (int v = 0; v < 8; ++v) {
                int row = m0b + wm * 64 + mt * 16 + half * 8 + v;
                C[(long)row * N + n0b + wn * 32 + nt * 16 + r] = c[mt][nt][v];
            }
        }
    }
}

// Fused causal depthwise conv (K=4) + sigmoid gating + chunked linear scan
// + SiLU gate. One block per (batch, head): 128 channels x 8 chunks of 512 steps.
__global__ __launch_bounds__(1024)
void LSTM_conv_scan(const float* __restrict__ y,
                    const float* __restrict__ cw,
                    float* __restrict__ out) {
    __shared__ float sA[8][HD];
    __shared__ float sB[8][HD];

    const int d     = threadIdx.x & (HD - 1);
    const int chunk = threadIdx.x >> 7;        // 0..7
    const int b     = blockIdx.x >> 4;
    const int g     = blockIdx.x & 15;

    const int colF = 2 * HID + g * (2 * HD) + d;
    const int colI = colF + HD;
    const int colH = HID + g * HD + d;
    const int colO = g * HD + d;

    const int cfb = (g * 2 * HD + d) * 4;
    const int cib = (g * 2 * HD + HD + d) * 4;
    const float cf0 = cw[cfb + 0], cf1 = cw[cfb + 1], cf2 = cw[cfb + 2], cf3 = cw[cfb + 3];
    const float ci0 = cw[cib + 0], ci1 = cw[cib + 1], ci2 = cw[cib + 2], ci3 = cw[cib + 3];

    const long rowBase = (long)b * SEQ * FOURH;
    const int  CH = SEQ / 8;
    const int  t0 = chunk * CH;

    // ---- pass 1 ----
    float wf0, wf1, wf2, wi0, wi1, wi2;
    {
        int t;
        t = t0 - 3; wf0 = (t >= 0) ? y[rowBase + (long)t * FOURH + colF] : 0.0f;
                    wi0 = (t >= 0) ? y[rowBase + (long)t * FOURH + colI] : 0.0f;
        t = t0 - 2; wf1 = (t >= 0) ? y[rowBase + (long)t * FOURH + colF] : 0.0f;
                    wi1 = (t >= 0) ? y[rowBase + (long)t * FOURH + colI] : 0.0f;
        t = t0 - 1; wf2 = (t >= 0) ? y[rowBase + (long)t * FOURH + colF] : 0.0f;
                    wi2 = (t >= 0) ? y[rowBase + (long)t * FOURH + colI] : 0.0f;
    }
    float Aacc = 1.0f, Bacc = 0.0f;
    for (int t = t0; t < t0 + CH; ++t) {
        long ro = rowBase + (long)t * FOURH;
        float nf = y[ro + colF], ni = y[ro + colI];
        float f = sigmoidf_dev(cf0 * wf0 + cf1 * wf1 + cf2 * wf2 + cf3 * nf);
        float i = sigmoidf_dev(ci0 * wi0 + ci1 * wi1 + ci2 * wi2 + ci3 * ni);
        float den = 1.0f / (f + i + 1e-4f);
        float a  = f * den;
        float bv = y[ro + colH] * i * den;
        Aacc = a * Aacc;
        Bacc = a * Bacc + bv;
        wf0 = wf1; wf1 = wf2; wf2 = nf;
        wi0 = wi1; wi1 = wi2; wi2 = ni;
    }
    sA[chunk][d] = Aacc;
    sB[chunk][d] = Bacc;
    __syncthreads();

    if (chunk == 0) {
        float h = 0.0f;
        #pragma unroll
        for (int cc = 0; cc < 8; ++cc) {
            float nh = sA[cc][d] * h + sB[cc][d];
            sB[cc][d] = h;
            h = nh;
        }
    }
    __syncthreads();
    float h = sB[chunk][d];

    // ---- pass 2 ----
    {
        int t;
        t = t0 - 3; wf0 = (t >= 0) ? y[rowBase + (long)t * FOURH + colF] : 0.0f;
                    wi0 = (t >= 0) ? y[rowBase + (long)t * FOURH + colI] : 0.0f;
        t = t0 - 2; wf1 = (t >= 0) ? y[rowBase + (long)t * FOURH + colF] : 0.0f;
                    wi1 = (t >= 0) ? y[rowBase + (long)t * FOURH + colI] : 0.0f;
        t = t0 - 1; wf2 = (t >= 0) ? y[rowBase + (long)t * FOURH + colF] : 0.0f;
                    wi2 = (t >= 0) ? y[rowBase + (long)t * FOURH + colI] : 0.0f;
    }
    for (int t = t0; t < t0 + CH; ++t) {
        long ro = rowBase + (long)t * FOURH;
        float nf = y[ro + colF], ni = y[ro + colI];
        float f = sigmoidf_dev(cf0 * wf0 + cf1 * wf1 + cf2 * wf2 + cf3 * nf);
        float i = sigmoidf_dev(ci0 * wi0 + ci1 * wi1 + ci2 * wi2 + ci3 * ni);
        float den = 1.0f / (f + i + 1e-4f);
        float a  = f * den;
        float bv = y[ro + colH] * i * den;
        h = a * h + bv;
        float o1 = y[ro + colO];
        float sl = o1 * sigmoidf_dev(o1);
        out[((long)b * SEQ + t) * HID + colO] = sl * h;
        wf0 = wf1; wf1 = wf2; wf2 = nf;
        wi0 = wi1; wi1 = wi2; wi2 = ni;
    }
}

extern "C" void kernel_launch(void* const* d_in, const int* in_sizes, int n_in,
                              void* d_out, int out_size, void* d_ws, size_t ws_size,
                              hipStream_t stream) {
    const float* x  = (const float*)d_in[0];   // [B,S,E]
    const float* w1 = (const float*)d_in[1];   // [4H,E]
    const float* w2 = (const float*)d_in[2];   // [E,H]
    const float* cw = (const float*)d_in[3];   // [2H,4]
    float* outp = (float*)d_out;               // [B,S,E]

    char* p = (char*)d_ws;
    float* y   = (float*)p;          p += (size_t)BSZ * SEQ * FOURH * 4;  // 512 MB
    float* mid = (float*)p;          p += (size_t)BSZ * SEQ * HID * 4;    // 128 MB
    unsigned short* xh  = (unsigned short*)p; p += (size_t)BSZ * SEQ * EMB * 2;
    unsigned short* xl  = (unsigned short*)p; p += (size_t)BSZ * SEQ * EMB * 2;
    unsigned short* w1h = (unsigned short*)p; p += (size_t)FOURH * EMB * 2;
    unsigned short* w1l = (unsigned short*)p; p += (size_t)FOURH * EMB * 2;
    unsigned short* oh  = (unsigned short*)p; p += (size_t)BSZ * SEQ * HID * 2;
    unsigned short* ol  = (unsigned short*)p; p += (size_t)BSZ * SEQ * HID * 2;
    unsigned short* w2h = (unsigned short*)p; p += (size_t)EMB * HID * 2;
    unsigned short* w2l = (unsigned short*)p; p += (size_t)EMB * HID * 2;

    const int M = BSZ * SEQ;

    LSTM_split_bf16<<<2048, 256, 0, stream>>>(x,  xh,  xl,  M * EMB);
    LSTM_split_bf16<<<2048, 256, 0, stream>>>(w1, w1h, w1l, FOURH * EMB);

    dim3 g1(FOURH / TN, M / TM);
    LSTM_gemm_lds<<<g1, 256, 0, stream>>>(xh, xl, w1h, w1l, y, M, FOURH, EMB);

    LSTM_conv_scan<<<BSZ * NHEAD, 1024, 0, stream>>>(y, cw, mid);

    LSTM_split_bf16<<<2048, 256, 0, stream>>>(mid, oh, ol, M * HID);
    LSTM_split_bf16<<<2048, 256, 0, stream>>>(w2,  w2h, w2l, EMB * HID);

    dim3 g2(EMB / TN, M / TM);
    LSTM_gemm_lds<<<g2, 256, 0, stream>>>(oh, ol, w2h, w2l, outp, M, EMB, HID);
}
